// InBatchNegativesSampler_12489764896910
// MI455X (gfx1250) — compile-verified, matrix operand order checked by reading, not run on hardware
//
#include <hip/hip_runtime.h>
#include <hip/hip_bf16.h>
#include <stdint.h>

typedef __attribute__((ext_vector_type(2)))  float    v2f;
typedef __attribute__((ext_vector_type(4)))  float    v4f;
typedef __attribute__((ext_vector_type(8)))  float    v8f;
typedef __attribute__((ext_vector_type(16))) _Float16 v16h;

#if __has_builtin(__builtin_amdgcn_wmma_f32_16x16x4_f32)
#define HAVE_WMMA_F32X4 1
#else
#define HAVE_WMMA_F32X4 0
#endif

// ---------------------------------------------------------------------------
// Kernel A: row inverse-norms via WMMA.
// 16 rows per wave. sum_k x[m,k]^2 = (X.^2)(16xK) * ones(Kx16), accumulated in
// the f32 C-matrix. After the K loop, lane (0|16) VGPR j holds the sum for row
// rowBase + (lane>=16 ? 8 : 0) + j  (all N columns identical since B == ones).
// ---------------------------------------------------------------------------
__global__ void __launch_bounds__(256) norm_wmma_kernel(
    const float* __restrict__ emb, float* __restrict__ inv_norm, int C, int D)
{
    const int lane   = threadIdx.x & 31;
    const int wave   = threadIdx.x >> 5;
    const int group  = blockIdx.x * (blockDim.x >> 5) + wave;
    const int groups = (C + 15) >> 4;
    if (group >= groups) return;          // uniform per wave: EXEC stays all-1s

    const int rowBase = group << 4;
    const int m  = lane & 15;             // matrix row owned by this lane
    const int kh = lane >> 4;             // which K-half this lane supplies
    int row = rowBase + m;
    if (row >= C) row = C - 1;            // clamp: duplicated rows never stored
    const float* rp = emb + (size_t)row * D;

    v8f acc = {};
#if HAVE_WMMA_F32X4
    v2f bones; bones.x = 1.0f; bones.y = 1.0f;
    for (int kb = 0; kb < D; kb += 4) {
        v2f x = *(const v2f*)(rp + kb + (kh << 1));   // 8B aligned
        v2f a; a.x = x.x * x.x; a.y = x.y * x.y;
        acc = __builtin_amdgcn_wmma_f32_16x16x4_f32(
            false, a, false, bones, (short)0, acc, false, false);
    }
#else
    v16h bones;
    #pragma unroll
    for (int i = 0; i < 16; ++i) bones[i] = (_Float16)1.0f;
    for (int kb = 0; kb < D; kb += 32) {
        v16h a;
        #pragma unroll
        for (int j = 0; j < 8; ++j) {
            float x0 = rp[kb + (kh << 3) + j];          // K = kh*8 + j
            float x1 = rp[kb + 16 + (kh << 3) + j];     // K = 16 + kh*8 + j
            a[j]     = (_Float16)(x0 * x0);
            a[8 + j] = (_Float16)(x1 * x1);
        }
        acc = __builtin_amdgcn_wmma_f32_16x16x32_f16(
            false, a, false, bones, (short)0, acc, false, false);
    }
#endif

    if ((lane & 15) == 0) {               // lanes 0 and 16 hold rows 0-7 / 8-15
        const int mb = rowBase + (kh << 3);
        #pragma unroll
        for (int j = 0; j < 8; ++j) {
            int r = mb + j;
            if (r < C) {
                float nrm = sqrtf(acc[j]);
                inv_norm[r] = 1.0f / fmaxf(nrm, 1e-6f);
            }
        }
    }
}

// ---------------------------------------------------------------------------
// Deterministic counter-based hash (splitmix32 flavor).
// ---------------------------------------------------------------------------
__device__ __forceinline__ uint32_t mix32(uint32_t x)
{
    x ^= x >> 16; x *= 0x21F0AAADu;
    x ^= x >> 15; x *= 0x735A2D97u;
    x ^= x >> 15;
    return x;
}

// ---------------------------------------------------------------------------
// Kernel B: one wave per (p, s) output row.
//  1) wave-parallel rejection sampling: 32 candidates/round, ballot, take the
//     first valid one (uniform over the valid set, E[rounds] ~ 1.01).
//  2) gather the 1KB row from the L2-resident table, scale by inv_norm (fused
//     l2-normalization), stream out with non-temporal 128b stores.
// ---------------------------------------------------------------------------
__global__ void __launch_bounds__(256) sample_gather_kernel(
    const int*   __restrict__ positive_ids,
    const int*   __restrict__ jagged_seq_ids,
    const int*   __restrict__ cached_ids,
    const int*   __restrict__ cached_seq_ids,
    const float* __restrict__ emb,
    const float* __restrict__ inv_norm,
    float*       __restrict__ out_ids,
    float*       __restrict__ out_emb,
    int C, int D, int S, long long tasks)
{
    const int lane = threadIdx.x & 31;
    const int wave = threadIdx.x >> 5;
    const long long task = (long long)blockIdx.x * (blockDim.x >> 5) + wave;
    if (task >= tasks) return;

    const int p   = (int)(task / S);
    const int jag = jagged_seq_ids[p];    // uniform across wave -> broadcast
    const int pid = positive_ids[p];

    const uint32_t tkey = (uint32_t)task * 0x9E3779B9u;

    int chosen = -1;
    for (int rnd = 0; rnd < 4 && chosen < 0; ++rnd) {
        uint32_t h = mix32(tkey ^ (uint32_t)(rnd * 32 + lane) * 0x85EBCA77u
                                ^ 0xC0FFEE11u);
        int c = (int)(((uint64_t)h * (uint32_t)C) >> 32);  // unbiased [0,C)
        bool ok = (cached_seq_ids[c] != jag) & (cached_ids[c] != pid);
        unsigned mm = (unsigned)__ballot(ok);              // wave32: low 32b
        if (mm) {
            int src = __builtin_ctz(mm);
            chosen  = __shfl(c, src, 32);
        }
    }
    if (chosen < 0) {   // fully-masked row fallback: uniform over all columns
        uint32_t h = mix32(tkey ^ 0xDEADBEEFu);
        chosen = (int)(((uint64_t)h * (uint32_t)C) >> 32);
    }

    if (lane == 0) out_ids[task] = (float)cached_ids[chosen];

    const float scale = inv_norm[chosen];
    const v4f* src = (const v4f*)(emb + (size_t)chosen * D);
    v4f*       dst = (v4f*)(out_emb + (size_t)task * D);
    const int nv4 = D >> 2;                         // D=256 -> 64 vec4
    for (int i = lane; i < nv4; i += 32) {
        v4f v = src[i];                             // L2 hit (16MB table)
        v *= scale;
        __builtin_nontemporal_store(v, dst + i);    // don't rinse L2
    }
}

extern "C" void kernel_launch(void* const* d_in, const int* in_sizes, int n_in,
                              void* d_out, int out_size, void* d_ws, size_t ws_size,
                              hipStream_t stream)
{
    (void)n_in; (void)ws_size;
    const int*   positive_ids   = (const int*)d_in[0];
    const int*   jagged_seq_ids = (const int*)d_in[1];
    const int*   cached_ids     = (const int*)d_in[2];
    const int*   cached_seq_ids = (const int*)d_in[3];
    const float* emb            = (const float*)d_in[4];
    // d_in[5] = num_to_sample lives on device; recover S from host-side sizes.

    const int P = in_sizes[0];
    const int C = in_sizes[2];
    const int D = in_sizes[4] / (C > 0 ? C : 1);
    const long long tasks = (long long)out_size / (long long)(D + 1);  // P*S
    const int S = (int)(tasks / (P > 0 ? P : 1));

    float* inv_norm = (float*)d_ws;                    // C floats (64 KB)
    float* out_ids  = (float*)d_out;                   // [P*S]
    float* out_emb  = (float*)d_out + tasks;           // [P*S*D]

    const int groups  = (C + 15) / 16;                 // 16 rows per wave
    const int blocksA = (groups + 7) / 8;              // 8 waves per block
    hipLaunchKernelGGL(norm_wmma_kernel, dim3(blocksA), dim3(256), 0, stream,
                       emb, inv_norm, C, D);

    const long long blocksB = (tasks + 7) / 8;         // 1 wave per (p,s)
    hipLaunchKernelGGL(sample_gather_kernel, dim3((unsigned)blocksB), dim3(256),
                       0, stream,
                       positive_ids, jagged_seq_ids, cached_ids, cached_seq_ids,
                       emb, inv_norm, out_ids, out_emb, C, D, S, tasks);
}
